// BaselineTargetHead_88673894793271
// MI455X (gfx1250) — matrix operand order repaired and checked by reading, not compiled, not run on hardware
//
#include <hip/hip_runtime.h>
#include <hip/hip_bf16.h>

typedef __attribute__((ext_vector_type(16))) __bf16 v16bf;
typedef __attribute__((ext_vector_type(8)))  __bf16 v8bf;
typedef __attribute__((ext_vector_type(8)))  float  v8f;

#define KPAD 40   // 32 K-elems + 8 pad (bf16) -> 80B row stride, spreads LDS banks; keeps 16B alignment

// Generic per-batch dynamic FC layer: Out[b,o,n] = sigmoid( sum_i W[b,o,i]*X[b,i,n] + Bias[b,o] )
// X: [B][I][64] f32, W: [B][O][I] f32, Bias: [B][O] f32, Out: [B][O][64] f32
// grid = (O/128, B), block = 256 (8 wave32). Wave w computes rows [w*16, w*16+16) x all 64 cols.
__global__ __launch_bounds__(256) void dynfc_wmma_kernel(
    const float* __restrict__ X,
    const float* __restrict__ W,
    const float* __restrict__ Bias,
    float* __restrict__ Out,
    int I, int O)
{
    const int b    = blockIdx.y;
    const int o0   = blockIdx.x * 128;
    const int t    = threadIdx.x;
    const int wave = t >> 5;
    const int lane = t & 31;
    const int lm   = lane & 15;   // row/col within tile
    const int half = lane >> 4;   // lane-half selects K sub-range

    // X chunk staged transposed (N-major) as bf16: xt[n*KPAD + k], k in [0,32)
    __shared__ alignas(16) __bf16 xt[64 * KPAD];

    // This lane supplies A-matrix row M = lm of the wave's 16-row tile.
    const float* __restrict__ Arow = W + ((size_t)b * O + (size_t)(o0 + wave * 16 + lm)) * I;
    const float* __restrict__ Xb   = X + (size_t)b * I * 64;

    const v8f zero = {0.f, 0.f, 0.f, 0.f, 0.f, 0.f, 0.f, 0.f};
    v8f acc[4];
    acc[0] = zero; acc[1] = zero; acc[2] = zero; acc[3] = zero;

    for (int kb = 0; kb < I; kb += 32) {
        __syncthreads();  // previous iteration's B reads done
        // ---- cooperative stage: X[b, kb..kb+31, 0..63] -> LDS transposed bf16 ----
        {
            const int e0 = t * 8;          // 256 threads * 8 elems = 32*64 chunk
            const int k  = e0 >> 6;        // local k row
            const int n0 = e0 & 63;        // starting column
            const float4 f0 = *(const float4*)(Xb + (size_t)(kb + k) * 64 + n0);
            const float4 f1 = *(const float4*)(Xb + (size_t)(kb + k) * 64 + n0 + 4);
            xt[(n0 + 0) * KPAD + k] = (__bf16)f0.x;
            xt[(n0 + 1) * KPAD + k] = (__bf16)f0.y;
            xt[(n0 + 2) * KPAD + k] = (__bf16)f0.z;
            xt[(n0 + 3) * KPAD + k] = (__bf16)f0.w;
            xt[(n0 + 4) * KPAD + k] = (__bf16)f1.x;
            xt[(n0 + 5) * KPAD + k] = (__bf16)f1.y;
            xt[(n0 + 6) * KPAD + k] = (__bf16)f1.z;
            xt[(n0 + 7) * KPAD + k] = (__bf16)f1.w;
        }
        __syncthreads();

        // ---- A fragment (16x32 bf16): lane holds K runs [8*half,8*half+8) and [16+8*half, +8) ----
        v16bf a;
        {
            const float* p = Arow + kb + half * 8;
            const float4 f0 = *(const float4*)(p);
            const float4 f1 = *(const float4*)(p + 4);
            const float4 f2 = *(const float4*)(p + 16);
            const float4 f3 = *(const float4*)(p + 20);
            a[0]  = (__bf16)f0.x; a[1]  = (__bf16)f0.y; a[2]  = (__bf16)f0.z; a[3]  = (__bf16)f0.w;
            a[4]  = (__bf16)f1.x; a[5]  = (__bf16)f1.y; a[6]  = (__bf16)f1.z; a[7]  = (__bf16)f1.w;
            a[8]  = (__bf16)f2.x; a[9]  = (__bf16)f2.y; a[10] = (__bf16)f2.z; a[11] = (__bf16)f2.w;
            a[12] = (__bf16)f3.x; a[13] = (__bf16)f3.y; a[14] = (__bf16)f3.z; a[15] = (__bf16)f3.w;
        }

        // ---- preload ALL four B fragments (32x16 bf16 each) so the 8 ds_load_b128
        //      are in flight together and waits can be staged, then issue 4 WMMAs ----
        v16bf bb[4];
        #pragma unroll
        for (int nt = 0; nt < 4; ++nt) {
            const int n = nt * 16 + lm;
            const v8bf b0 = *(const v8bf*)&xt[n * KPAD + half * 8];
            const v8bf b1 = *(const v8bf*)&xt[n * KPAD + half * 8 + 16];
            #pragma unroll
            for (int j = 0; j < 8; ++j) { bb[nt][j] = b0[j]; bb[nt][j + 8] = b1[j]; }
        }
        #pragma unroll
        for (int nt = 0; nt < 4; ++nt) {
            acc[nt] = __builtin_amdgcn_wmma_f32_16x16x32_bf16(
                false, a, false, bb[nt], (short)0, acc[nt], false, false);
        }
    }

    // ---- epilogue: bias + sigmoid + store (C/D layout: VGPR r -> M = r + 8*half, lane -> col) ----
    const float* __restrict__ biasp = Bias + (size_t)b * O;
    float* __restrict__ outp = Out + (size_t)b * O * 64;
    #pragma unroll
    for (int r = 0; r < 8; ++r) {
        const int grow = o0 + wave * 16 + r + half * 8;
        const float bv = biasp[grow];
        #pragma unroll
        for (int nt = 0; nt < 4; ++nt) {
            float v = acc[nt][r] + bv;
            v = 1.0f / (1.0f + __expf(-v));
            outp[(size_t)grow * 64 + nt * 16 + lm] = v;
        }
    }
}

// Final layer: O = 1, no activation. out[b,n] = dot(W5[b,:], X[b,:,n]) + B5[b]
__global__ __launch_bounds__(64) void fc5_kernel(
    const float* __restrict__ X,    // [B][128][64]
    const float* __restrict__ W5,   // [B][1][128]
    const float* __restrict__ B5,   // [B][1]
    float* __restrict__ out)        // [B][64]
{
    const int b = blockIdx.x;
    const int n = threadIdx.x;
    const float* __restrict__ w = W5 + (size_t)b * 128;   // uniform -> scalar loads
    const float* __restrict__ x = X + (size_t)b * 128 * 64;
    float acc = 0.f;
    #pragma unroll 8
    for (int i = 0; i < 128; ++i)
        acc += w[i] * x[(size_t)i * 64 + n];
    out[b * 64 + n] = acc + B5[b];
}

extern "C" void kernel_launch(void* const* d_in, const int* in_sizes, int n_in,
                              void* d_out, int out_size, void* d_ws, size_t ws_size,
                              hipStream_t stream) {
    const float* x0 = (const float*)d_in[0];
    const float* w1 = (const float*)d_in[1];  const float* b1 = (const float*)d_in[2];
    const float* w2 = (const float*)d_in[3];  const float* b2 = (const float*)d_in[4];
    const float* w3 = (const float*)d_in[5];  const float* b3 = (const float*)d_in[6];
    const float* w4 = (const float*)d_in[7];  const float* b4 = (const float*)d_in[8];
    const float* w5 = (const float*)d_in[9];  const float* b5 = (const float*)d_in[10];

    float* ws = (float*)d_ws;
    const size_t SLOT = (size_t)64 * 1024 * 64;   // largest intermediate (out1), 16 MB
    float* o1 = ws;          // [64][1024][64]
    float* o2 = ws + SLOT;   // [64][512][64]
    float* o3 = ws;          // [64][256][64]  (reuses o1, dead after L2; stream-ordered)
    float* o4 = ws + SLOT;   // [64][128][64]  (reuses o2, dead after L3)

    dim3 blk(256);
    dynfc_wmma_kernel<<<dim3(8, 64), blk, 0, stream>>>(x0, w1, b1, o1, 2048, 1024);
    dynfc_wmma_kernel<<<dim3(4, 64), blk, 0, stream>>>(o1, w2, b2, o2, 1024, 512);
    dynfc_wmma_kernel<<<dim3(2, 64), blk, 0, stream>>>(o2, w3, b3, o3, 512, 256);
    dynfc_wmma_kernel<<<dim3(1, 64), blk, 0, stream>>>(o3, w4, b4, o4, 256, 128);
    fc5_kernel<<<dim3(64), dim3(64), 0, stream>>>(o4, w5, b5, (float*)d_out);
}
